// DKVMN_979252543889
// MI455X (gfx1250) — matrix operand order repaired and checked by reading, not compiled, not run on hardware
//
#include <hip/hip_runtime.h>
#include <hip/hip_bf16.h>

typedef __attribute__((ext_vector_type(16))) _Float16 v16h;
typedef __attribute__((ext_vector_type(8)))  float    v8f;

#define KCNT   50000
#define DK     64
#define NSLOT  64
#define HDIM   64
#define VDIM   128
#define FTN    50
#define BSZ    2048
#define TSZ    128
#define TILE_B 16
#define NTHR   256

// LDS layout (bytes):
//   hS    : float [16][64][64]          = 262144
//   catA  : _Float16 [16][128]          =   4096  (cols 0..63 = rt, 64..127 = kn)
//   vtA   : _Float16 [16][128]          =   4096
//   alphaS: float [16][64]              =   4096
//   ftvS  : float [16][64]              =   4096
//   etS   : float [16][64]              =   4096
//   atS   : float [16][64]              =   4096
#define SMEM_BYTES (262144 + 4096*6)

__device__ __forceinline__ v8f wmma_f16(v16h a, v16h b, v8f c) {
  return __builtin_amdgcn_wmma_f32_16x16x32_f16(false, a, false, b, (short)0, c,
                                                false, false);
}

// Fast transcendentals on the step critical path: v_exp_f32 + v_rcp_f32,
// no IEEE div_scale/div_fmas sequences.
__device__ __forceinline__ float fast_sigmoid(float x) {
  return __builtin_amdgcn_rcpf(1.f + __expf(-x));
}
__device__ __forceinline__ float fast_tanh(float x) {
  // tanh(x) = 1 - 2/(e^{2x}+1); rcp(inf)=0 gives correct +/-1 saturation
  return 1.f - 2.f * __builtin_amdgcn_rcpf(__expf(2.f * x) + 1.f);
}

// Load A tile [16 x 32] (f16) starting at column k0 from a row-major buffer
// with row stride 128 halves.  A layout (contiguous K, matches B):
//   lane L: M = L&15, kbase = (L>>4)*16
//   half i -> K = kbase + i
__device__ __forceinline__ v16h load_a_tile(const _Float16* base, int lane, int k0) {
  const int m  = lane & 15;
  const int kb = (lane >> 4) << 4;
  v16h a;
#pragma unroll
  for (int i = 0; i < 16; ++i) {
    const int krel = kb + i;
    a[i] = base[m * 128 + k0 + krel];
  }
  return a;
}

__global__ __launch_bounds__(NTHR)
void dkvmn_scan_kernel(const int* __restrict__ q_idx, const float* __restrict__ score,
                       const float* __restrict__ km,    const float* __restrict__ h_init,
                       const float* __restrict__ W_k,   const float* __restrict__ b_k,
                       const float* __restrict__ W_ft,  const float* __restrict__ b_ft,
                       const float* __restrict__ W_s,   const float* __restrict__ b_s,
                       const float* __restrict__ W_cks, const float* __restrict__ b_cks,
                       const float* __restrict__ W_e,   const float* __restrict__ b_e,
                       const float* __restrict__ W_a,   const float* __restrict__ b_a,
                       float* __restrict__ out)
{
  extern __shared__ char smem[];
  float*    hS     = (float*)smem;                               // 16*64*64
  _Float16* catA   = (_Float16*)(smem + 262144);                 // 16*128
  _Float16* vtA    = (_Float16*)(smem + 262144 + 4096);          // 16*128
  float*    alphaS = (float*)(smem + 262144 + 8192);             // 16*64
  float*    ftvS   = alphaS + TILE_B * 64;
  float*    etS    = ftvS   + TILE_B * 64;
  float*    atS    = etS    + TILE_B * 64;

  const int tid  = threadIdx.x;
  const int lane = tid & 31;
  const int wid  = tid >> 5;          // 8 waves: 0-3 = scores/ft chain, 4-7 = et/at chain
  const int b0   = blockIdx.x * TILE_B;

  const int nloc = lane & 15;
  const int kblk = lane >> 4;         // K-block-of-16 held by this lane (B operand)
  const int ncol = (wid & 3) * 16 + nloc;  // output column of this lane's GEMM tile
  const int mrow = (lane >> 4) * 8;   // C/D: M = mrow + r

  // ---- init h state: broadcast h_initial to all 16 batch rows (float4) ----
  {
    const float4* src = (const float4*)h_init;   // 1024 float4
    float4* dst = (float4*)hS;                   // 16384 float4
    for (int e = tid; e < TILE_B * 1024; e += NTHR)
      dst[e] = src[e & 1023];
  }

  // ---- f16 B-operands in registers (per-wave-half union), reused all 128 steps ----
  // B layout: half i of lane L -> B[(L>>4)*16 + i][L&15]
  // wid<4 : Breg[0..1] = kmT k-tiles,  Breg[2..5] = W_ft k-tiles
  // wid>=4: Breg[0..3] = W_e  k-tiles, Breg[4..7] = W_a  k-tiles
  v16h Breg[8] = {};
  float bias0 = 0.f, bias1 = 0.f;
  if (wid < 4) {
#pragma unroll
    for (int kt = 0; kt < 2; ++kt)
#pragma unroll
      for (int i = 0; i < 16; ++i) {
        const int d = kt * 32 + kblk * 16 + i;        // B[d][n] = km[n][d]
        Breg[kt][i] = (_Float16)km[ncol * DK + d];
      }
#pragma unroll
    for (int kt = 0; kt < 4; ++kt)
#pragma unroll
      for (int i = 0; i < 16; ++i) {
        const int k = kt * 32 + kblk * 16 + i;
        Breg[2 + kt][i] = (ncol < FTN) ? (_Float16)W_ft[k * FTN + ncol] : (_Float16)0.f;
      }
    bias0 = (ncol < FTN) ? b_ft[ncol] : 0.f;
  } else {
#pragma unroll
    for (int kt = 0; kt < 4; ++kt)
#pragma unroll
      for (int i = 0; i < 16; ++i) {
        const int k = kt * 32 + kblk * 16 + i;
        Breg[kt][i]     = (_Float16)W_e[k * HDIM + ncol];
        Breg[4 + kt][i] = (_Float16)W_a[k * HDIM + ncol];
      }
    bias0 = b_e[ncol];
    bias1 = b_a[ncol];
  }
  const float bsc = b_s[0];

  __syncthreads();

  for (int t = 0; t < TSZ; ++t) {
    // ---- Phase A: gather kn -> catA[:,64:128], vt -> vtA (f16, bias folded) ----
    for (int e = tid; e < TILE_B * DK; e += NTHR) {
      const int r = e >> 6, d = e & 63;
      const int idx = q_idx[(b0 + r) * TSZ + t];
      catA[r * 128 + 64 + d] = (_Float16)(W_k[idx * DK + d] + b_k[d]);
    }
    for (int e = tid; e < TILE_B * VDIM; e += NTHR) {
      const int r = e >> 7, v = e & 127;
      const int idx = q_idx[(b0 + r) * TSZ + t];
      const float sc = score[(b0 + r) * TSZ + t];
      vtA[r * 128 + v] =
          (_Float16)(W_cks[idx * VDIM + v] + sc * W_cks[(idx + KCNT) * VDIM + v] + b_cks[v]);
    }
    __syncthreads();

    // ---- Phase B: waves 0-3 scores GEMM | waves 4-7 et GEMM (independent) ----
    if (wid < 4) {
      v8f c = {};
#pragma unroll
      for (int kt = 0; kt < 2; ++kt) {
        v16h a = load_a_tile(catA, lane, 64 + kt * 32);
        c = wmma_f16(a, Breg[kt], c);
      }
#pragma unroll
      for (int r = 0; r < 8; ++r)
        alphaS[(mrow + r) * 64 + ncol] = c[r];
    } else {
      v8f ce = {};
#pragma unroll
      for (int kt = 0; kt < 4; ++kt) {
        v16h a = load_a_tile(vtA, lane, kt * 32);
        ce = wmma_f16(a, Breg[kt], ce);
      }
#pragma unroll
      for (int r = 0; r < 8; ++r)
        etS[(mrow + r) * 64 + ncol] = fast_sigmoid(ce[r] + bias0);
    }
    __syncthreads();

    // ---- Phase C: waves 0-3 softmax | waves 4-7 at GEMM (independent) ----
    if (tid < 128) {
      const int row = tid >> 3, g = tid & 7;
      float v[8], mx = -1e30f;
#pragma unroll
      for (int j = 0; j < 8; ++j) { v[j] = alphaS[row * 64 + g * 8 + j]; mx = fmaxf(mx, v[j]); }
      mx = fmaxf(mx, __shfl_xor(mx, 1, 8));
      mx = fmaxf(mx, __shfl_xor(mx, 2, 8));
      mx = fmaxf(mx, __shfl_xor(mx, 4, 8));
      float s = 0.f;
#pragma unroll
      for (int j = 0; j < 8; ++j) { v[j] = __expf(v[j] - mx); s += v[j]; }
      s += __shfl_xor(s, 1, 8);
      s += __shfl_xor(s, 2, 8);
      s += __shfl_xor(s, 4, 8);
      const float inv = __builtin_amdgcn_rcpf(s);
#pragma unroll
      for (int j = 0; j < 8; ++j) alphaS[row * 64 + g * 8 + j] = v[j] * inv;
    } else {
      v8f ca = {};
#pragma unroll
      for (int kt = 0; kt < 4; ++kt) {
        v16h a = load_a_tile(vtA, lane, kt * 32);
        ca = wmma_f16(a, Breg[4 + kt], ca);
      }
#pragma unroll
      for (int r = 0; r < 8; ++r)
        atS[(mrow + r) * 64 + ncol] = fast_tanh(ca[r] + bias1);
    }
    __syncthreads();

    // ---- Phase D: rt[b,:] = alpha[b,:] @ h[b]  ->  catA[:,0:64]  (all 256) ----
    {
      const int b = tid >> 4, hb = (tid & 15) * 4;
      float acc0 = 0.f, acc1 = 0.f, acc2 = 0.f, acc3 = 0.f;
      for (int n = 0; n < 64; ++n) {
        const float a = alphaS[b * 64 + n];
        const float4 x = *(const float4*)(hS + b * 4096 + n * 64 + hb);
        acc0 += a * x.x; acc1 += a * x.y; acc2 += a * x.z; acc3 += a * x.w;
      }
      catA[b * 128 + hb + 0] = (_Float16)acc0;
      catA[b * 128 + hb + 1] = (_Float16)acc1;
      catA[b * 128 + hb + 2] = (_Float16)acc2;
      catA[b * 128 + hb + 3] = (_Float16)acc3;
    }
    __syncthreads();

    // ---- Phase E: waves 0-3 ft GEMM | waves 4-7 prefetch step t+1 gathers ----
    if (wid < 4) {
      v8f cf = {};
#pragma unroll
      for (int kt = 0; kt < 4; ++kt) {
        v16h a = load_a_tile(catA, lane, kt * 32);
        cf = wmma_f16(a, Breg[2 + kt], cf);
      }
#pragma unroll
      for (int r = 0; r < 8; ++r)
        ftvS[(mrow + r) * 64 + ncol] = (ncol < FTN) ? fast_tanh(cf[r] + bias0) : 0.f;
    } else if (t + 1 < TSZ) {
      // one cacheline (128B) per prefetch; 8 threads cover each batch row
      const int p = tid - 128;              // 0..127
      const int r = p >> 3, seg = p & 7;    // 16 rows x 8 segments
      const int idx = q_idx[(b0 + r) * TSZ + t + 1];
      const float* cks = W_cks + (size_t)(idx + (seg >= 4 ? KCNT : 0)) * VDIM + (seg & 3) * 32;
      __builtin_prefetch(cks, 0, 3);        // global_prefetch_b8
      if (seg < 2)
        __builtin_prefetch(W_k + (size_t)idx * DK + seg * 32, 0, 3);
    }
    __syncthreads();

    // ---- Phase F: pred (threads 0-15) + erase/add h update (all 256) ----
    if (tid < TILE_B) {
      float s = bsc;
      for (int n = 0; n < FTN; ++n) s += ftvS[tid * 64 + n] * W_s[n];
      out[(size_t)(b0 + tid) * TSZ + t] = fast_sigmoid(s);
    }
    {
      const int b = tid >> 4, hb = (tid & 15) * 4;
      float e0 = etS[b * 64 + hb + 0], e1 = etS[b * 64 + hb + 1];
      float e2 = etS[b * 64 + hb + 2], e3 = etS[b * 64 + hb + 3];
      float a0 = atS[b * 64 + hb + 0], a1 = atS[b * 64 + hb + 1];
      float a2 = atS[b * 64 + hb + 2], a3 = atS[b * 64 + hb + 3];
      for (int n = 0; n < 64; ++n) {
        const float al = alphaS[b * 64 + n];
        float4 x = *(float4*)(hS + b * 4096 + n * 64 + hb);
        x.x = x.x * (1.f - al * e0) + al * a0;
        x.y = x.y * (1.f - al * e1) + al * a1;
        x.z = x.z * (1.f - al * e2) + al * a2;
        x.w = x.w * (1.f - al * e3) + al * a3;
        *(float4*)(hS + b * 4096 + n * 64 + hb) = x;
      }
    }
    __syncthreads();
  }

  // ---- write h_final ----
  {
    const int b = tid >> 4, hb = (tid & 15) * 4;
    float* dst = out + (size_t)BSZ * TSZ + (size_t)(b0 + b) * NSLOT * HDIM;
    for (int n = 0; n < 64; ++n)
      *(float4*)(dst + n * 64 + hb) = *(const float4*)(hS + b * 4096 + n * 64 + hb);
  }
}

extern "C" void kernel_launch(void* const* d_in, const int* in_sizes, int n_in,
                              void* d_out, int out_size, void* d_ws, size_t ws_size,
                              hipStream_t stream) {
  (void)in_sizes; (void)n_in; (void)out_size; (void)d_ws; (void)ws_size;

  const int*   q_idx  = (const int*)d_in[0];
  const float* score  = (const float*)d_in[1];
  const float* km     = (const float*)d_in[2];
  const float* h_init = (const float*)d_in[3];
  const float* W_k    = (const float*)d_in[4];
  const float* b_k    = (const float*)d_in[5];
  const float* W_ft   = (const float*)d_in[6];
  const float* b_ft   = (const float*)d_in[7];
  const float* W_s    = (const float*)d_in[8];
  const float* b_s    = (const float*)d_in[9];
  const float* W_cks  = (const float*)d_in[10];
  const float* b_cks  = (const float*)d_in[11];
  const float* W_e    = (const float*)d_in[12];
  const float* b_e    = (const float*)d_in[13];
  const float* W_a    = (const float*)d_in[14];
  const float* b_a    = (const float*)d_in[15];
  float* out = (float*)d_out;

  // CDNA5 allows up to 320KB LDS per workgroup; opt in to >64KB dynamic LDS.
  static_assert(SMEM_BYTES <= 320 * 1024, "LDS budget");
  hipFuncSetAttribute((const void*)dkvmn_scan_kernel,
                      hipFuncAttributeMaxDynamicSharedMemorySize, SMEM_BYTES);

  dim3 grid(BSZ / TILE_B);   // 128 workgroups, one 16-batch tile each
  dim3 block(NTHR);          // 8 waves: dual-chain dataflow split
  dkvmn_scan_kernel<<<grid, block, SMEM_BYTES, stream>>>(
      q_idx, score, km, h_init, W_k, b_k, W_ft, b_ft, W_s, b_s,
      W_cks, b_cks, W_e, b_e, W_a, b_a, out);
}
